// RotAttentionPool2d_58703613002516
// MI455X (gfx1250) — matrix operand order repaired and checked by reading, not compiled, not run on hardware
//
#include <hip/hip_runtime.h>
#include <hip/hip_bf16.h>
#include <cmath>

typedef __attribute__((ext_vector_type(16))) _Float16 v16h;
typedef __attribute__((ext_vector_type(8)))  float    v8f;
typedef int v4i_ __attribute__((vector_size(16)));
typedef __attribute__((address_space(1))) v4i_ g_v4i;   // global int4
typedef __attribute__((address_space(3))) v4i_ l_v4i;   // LDS int4

static constexpr int Bn   = 32;    // batch
static constexpr int Cc   = 768;   // channels / embed
static constexpr int Nsp  = 576;   // spatial tokens (24*24)
static constexpr int NTOK = 577;   // +cls
static constexpr int NPAD = 592;   // 37*16, padded token rows
static constexpr int EMB  = 768;
static constexpr int HD   = 96;
static constexpr int NHd  = 8;
static constexpr int NB_  = 24;    // rope bands

__device__ inline float waveSum(float v) {
#pragma unroll
  for (int o = 16; o; o >>= 1) v += __shfl_xor(v, o);
  return v;
}

// ---- K0: per-(b,c) mean over spatial -> t_cls (f32), Th cls row (f16), zero pads
__global__ __launch_bounds__(256) void k_mean(const float* __restrict__ x,
                                              float* __restrict__ t_cls,
                                              _Float16* __restrict__ Th) {
  int tid = threadIdx.x;
  int row = blockIdx.x * 8 + (tid >> 5);   // row = b*768 + c
  int lane = tid & 31;
  const float* xr = x + (size_t)row * Nsp;
  float s = 0.f;
#pragma unroll
  for (int i = 0; i < 18; ++i) s += xr[lane + 32 * i];
  s = waveSum(s);
  if (lane == 0) {
    float m = s * (1.0f / 576.0f);
    t_cls[row] = m;
    int b = row / Cc, c = row % Cc;
    _Float16* tb = Th + (size_t)b * NPAD * Cc;
    tb[c] = (_Float16)m;                                // token 0 = cls
    for (int j = 0; j < NPAD - NTOK; ++j)               // zero padding rows
      tb[(size_t)(NTOK + j) * Cc + c] = (_Float16)0.f;
  }
}

// ---- K1: transpose x (B,C,N) -> Th (B, 1+N, C) in f16 via LDS tiles
__global__ __launch_bounds__(256) void k_pack(const float* __restrict__ x,
                                              _Float16* __restrict__ Th) {
  __shared__ float tile[32][33];
  int b = blockIdx.z;
  int n0 = blockIdx.x * 32;
  int c0 = blockIdx.y * 32;
  int tid = threadIdx.x;
  int tx = tid & 31, ty = tid >> 5;
#pragma unroll
  for (int i = 0; i < 4; ++i) {
    int c = c0 + ty + 8 * i;
    tile[ty + 8 * i][tx] = x[((size_t)(b * Cc + c)) * Nsp + n0 + tx];
  }
  __syncthreads();
#pragma unroll
  for (int i = 0; i < 4; ++i) {
    int n = n0 + ty + 8 * i;
    Th[((size_t)b * NPAD + 1 + n) * Cc + c0 + tx] = (_Float16)tile[tx][ty + 8 * i];
  }
}

// ---- K2: Wk (rows 768..1535 of qkv_w) -> f16
__global__ __launch_bounds__(256) void k_cvt_wk(const float* __restrict__ qkv_w,
                                                _Float16* __restrict__ Wk_h) {
  int i = blockIdx.x * 256 + threadIdx.x;
  Wk_h[i] = (_Float16)qkv_w[(size_t)EMB * Cc + i];
}

// ---- K3: q for the CLS token only (f32)
__global__ __launch_bounds__(256) void k_qcls(const float* __restrict__ t_cls,
                                              const float* __restrict__ qkv_w,
                                              const float* __restrict__ qkv_b,
                                              float* __restrict__ q_cls) {
  int tid = threadIdx.x;
  int o = blockIdx.x * 8 + (tid >> 5);
  int lane = tid & 31;
  int b = o / EMB, e = o % EMB;
  const float* t = t_cls + (size_t)b * Cc;
  const float* w = qkv_w + (size_t)e * Cc;
  float s = 0.f;
  for (int c = lane; c < Cc; c += 32) s += t[c] * w[c];
  s = waveSum(s);
  if (lane == 0) q_cls[o] = s + qkv_b[e];
}

// RoPE + store epilogue for one 16x16 accumulator tile (all lanes execute).
__device__ inline void rope_store(v8f acc, int e, int b, int n0, int g,
                                  const float* __restrict__ qkv_b,
                                  float* __restrict__ K_rot) {
  float bias = qkv_b[EMB + e];
  int d = e % HD;
  int pair = d >> 1;
  int axis = pair / NB_;
  int jj = pair % NB_;
  float band = exp2f((3.0f * jj) * (1.0f / 23.0f));
#pragma unroll
  for (int r = 0; r < 8; ++r) {
    float v = acc[r] + bias;
    float partner = __shfl_xor(v, 1);    // column e^1 (rope rotate pair)
    int n = n0 + r + 8 * g;              // token index of this element
    float out = v;
    if (n != 0) {                        // cls token skips rope
      int p = n - 1;
      int pos = (axis == 0) ? (p / 24) : (p % 24);
      float coord = -1.0f + (2.0f / 23.0f) * (float)pos;
      float th = coord * 3.14159265358979323846f * band;
      float cs = __cosf(th), sn = __sinf(th);
      out = (d & 1) ? (cs * v + sn * partner) : (cs * v - sn * partner);
    }
    if (n < NTOK)
      K_rot[((size_t)b * NPAD + n) * Cc + e] = out;
  }
}

// ---- K4: K = Th @ Wk^T + bk, RoPE epilogue, f32 out.  WMMA f32_16x16x32_f16.
__global__ __launch_bounds__(256) void k_kgemm(const _Float16* __restrict__ Th,
                                               const _Float16* __restrict__ Wk_h,
                                               const float* __restrict__ qkv_b,
                                               float* __restrict__ K_rot) {
  __shared__ __align__(16) _Float16 sA[16 * Cc];   // 24 KB: 16 token rows
  int b  = blockIdx.y;
  int n0 = blockIdx.x * 16;
  int tid = threadIdx.x;

  // A tile is 16 consecutive rows of Th -> one contiguous 24576B chunk.
  const char* src = (const char*)(Th + ((size_t)b * NPAD + n0) * Cc);
#if __has_builtin(__builtin_amdgcn_global_load_async_to_lds_b128)
  {
    char* ldst = (char*)sA;
#pragma unroll
    for (int i = 0; i < 6; ++i) {
      int off = (tid + 256 * i) * 16;
      __builtin_amdgcn_global_load_async_to_lds_b128(
          (g_v4i*)(uintptr_t)(src + off),
          (l_v4i*)(uintptr_t)(ldst + off),
          0, 0);
    }
  }
#if __has_builtin(__builtin_amdgcn_s_wait_asynccnt)
  __builtin_amdgcn_s_wait_asynccnt(0);
#else
  asm volatile("s_wait_asynccnt 0" ::: "memory");
#endif
#else
  {
    const float4* s4 = (const float4*)src;
    float4* d4 = (float4*)sA;
#pragma unroll
    for (int i = 0; i < 6; ++i) d4[tid + 256 * i] = s4[tid + 256 * i];
  }
#endif
  __syncthreads();

  int wave = tid >> 5, lane = tid & 31;
  int g = (lane >> 4) & 1;        // lane group (ISA A/B fragment layout)
  int nloc = lane & 15;

  union F { v16h v; float4 q[2]; };
  // Each wave owns 96 output columns: 3 pairs of 16-col tiles.
  for (int jp = 0; jp < 3; ++jp) {
    int e0 = wave * 96 + jp * 32 + nloc;   // this lane's column in tile 0
    int e1 = e0 + 16;                      // ... and in tile 1
    v8f acc0 = {}, acc1 = {};
    const _Float16* pA  = sA + (size_t)nloc * Cc + 8 * g;  // A: K = 8g+i, 16+8g+i
    const _Float16* pB0 = Wk_h + (size_t)e0 * Cc + 8 * g;  // B: K = 8g+i, 16+8g+i
    const _Float16* pB1 = Wk_h + (size_t)e1 * Cc + 8 * g;
    F a[2], b0[2], b1[2];
    a[0].q[0]  = *(const float4*)(pA);       a[0].q[1]  = *(const float4*)(pA + 16);
    b0[0].q[0] = *(const float4*)(pB0);      b0[0].q[1] = *(const float4*)(pB0 + 16);
    b1[0].q[0] = *(const float4*)(pB1);      b1[0].q[1] = *(const float4*)(pB1 + 16);
#pragma unroll 4
    for (int kk = 0; kk < 24; ++kk) {
      int cur = kk & 1, nxt = cur ^ 1;
      if (kk + 1 < 24) {                    // software pipeline: fetch k+1 frags
        int k0 = (kk + 1) * 32;
        a[nxt].q[0]  = *(const float4*)(pA + k0);
        a[nxt].q[1]  = *(const float4*)(pA + k0 + 16);
        b0[nxt].q[0] = *(const float4*)(pB0 + k0);
        b0[nxt].q[1] = *(const float4*)(pB0 + k0 + 16);
        b1[nxt].q[0] = *(const float4*)(pB1 + k0);
        b1[nxt].q[1] = *(const float4*)(pB1 + k0 + 16);
        __builtin_prefetch(pB0 + k0 + 32, 0, 3);
      }
      // one A fragment feeds two independent accumulators
      acc0 = __builtin_amdgcn_wmma_f32_16x16x32_f16(false, a[cur].v, false, b0[cur].v,
                                                    (short)0, acc0, false, false);
      acc1 = __builtin_amdgcn_wmma_f32_16x16x32_f16(false, a[cur].v, false, b1[cur].v,
                                                    (short)0, acc1, false, false);
    }
    rope_store(acc0, e0, b, n0, g, qkv_b, K_rot);
    rope_store(acc1, e1, b, n0, g, qkv_b, K_rot);
  }
}

// ---- K5: per (b,h): logits -> softmax -> weighted token pooling -> Wv head proj
__global__ __launch_bounds__(256) void k_attn(const float* __restrict__ q_cls,
                                              const float* __restrict__ K_rot,
                                              const float* __restrict__ x,
                                              const float* __restrict__ t_cls,
                                              const float* __restrict__ qkv_w,
                                              const float* __restrict__ qkv_b,
                                              float* __restrict__ o_buf) {
  __shared__ float qs[HD];
  __shared__ float lg[NTOK];
  __shared__ float wt[Cc];
  __shared__ float red[256];
  int h = blockIdx.x, b = blockIdx.y, tid = threadIdx.x;
  if (tid < HD) qs[tid] = q_cls[(size_t)b * EMB + h * HD + tid];
  __syncthreads();
  const float scale = rsqrtf((float)HD);
  for (int n = tid; n < NTOK; n += 256) {
    const float* kr = K_rot + ((size_t)b * NPAD + n) * Cc + h * HD;
    float d = 0.f;
#pragma unroll 8
    for (int i = 0; i < HD; ++i) d += kr[i] * qs[i];
    lg[n] = d * scale;
  }
  __syncthreads();
  float m = -1e30f;
  for (int n = tid; n < NTOK; n += 256) m = fmaxf(m, lg[n]);
  red[tid] = m; __syncthreads();
  for (int s = 128; s; s >>= 1) { if (tid < s) red[tid] = fmaxf(red[tid], red[tid + s]); __syncthreads(); }
  m = red[0]; __syncthreads();
  float ssum = 0.f;
  for (int n = tid; n < NTOK; n += 256) { float e = __expf(lg[n] - m); lg[n] = e; ssum += e; }
  red[tid] = ssum; __syncthreads();
  for (int s = 128; s; s >>= 1) { if (tid < s) red[tid] += red[tid + s]; __syncthreads(); }
  float inv = 1.0f / red[0];
  __syncthreads();
  // attention-weighted token pooling (replaces V GEMM): wt[c] = sum_n attn_n * t[n,c]
  for (int c = tid; c < Cc; c += 256) {
    float a = lg[0] * t_cls[(size_t)b * Cc + c];
    const float* xr = x + ((size_t)b * Cc + c) * Nsp;
    for (int p = 0; p < Nsp; ++p) a += lg[p + 1] * xr[p];
    wt[c] = a * inv;
  }
  __syncthreads();
  int wave = tid >> 5, lane = tid & 31;
  for (int d0 = wave; d0 < HD; d0 += 8) {
    int e = 2 * EMB + h * HD + d0;               // Wv row
    const float* w = qkv_w + (size_t)e * Cc;
    float s = 0.f;
    for (int c = lane; c < Cc; c += 32) s += wt[c] * w[c];
    s = waveSum(s);
    if (lane == 0) o_buf[(size_t)b * EMB + h * HD + d0] = s + qkv_b[e];
  }
}

// ---- K6: output projection (CLS row only)
__global__ __launch_bounds__(256) void k_proj(const float* __restrict__ o_buf,
                                              const float* __restrict__ proj_w,
                                              const float* __restrict__ proj_b,
                                              float* __restrict__ out) {
  int tid = threadIdx.x;
  int o = blockIdx.x * 8 + (tid >> 5);
  int lane = tid & 31;
  int b = o / EMB, e = o % EMB;
  const float* ob = o_buf + (size_t)b * EMB;
  const float* w = proj_w + (size_t)e * Cc;
  float s = 0.f;
  for (int c = lane; c < Cc; c += 32) s += ob[c] * w[c];
  s = waveSum(s);
  if (lane == 0) out[o] = s + proj_b[e];
}

extern "C" void kernel_launch(void* const* d_in, const int* in_sizes, int n_in,
                              void* d_out, int out_size, void* d_ws, size_t ws_size,
                              hipStream_t stream) {
  (void)in_sizes; (void)n_in; (void)out_size; (void)ws_size;
  const float* x      = (const float*)d_in[0];
  const float* qkv_w  = (const float*)d_in[1];
  const float* qkv_b  = (const float*)d_in[2];
  const float* proj_w = (const float*)d_in[3];
  const float* proj_b = (const float*)d_in[4];
  float* out = (float*)d_out;

  // workspace layout (~88.8 MB)
  float* t_cls = (float*)d_ws;                       // 32*768 f32
  float* q_cls = t_cls + 24576;                      // 32*768 f32
  float* o_buf = q_cls + 24576;                      // 32*768 f32
  _Float16* Wk_h = (_Float16*)(o_buf + 24576);       // 768*768 f16
  _Float16* Th   = Wk_h + 589824;                    // 32*592*768 f16
  float* K_rot   = (float*)(Th + (size_t)Bn * NPAD * Cc); // 32*592*768 f32

  k_mean <<<3072, 256, 0, stream>>>(x, t_cls, Th);
  k_pack <<<dim3(18, 24, 32), 256, 0, stream>>>(x, Th);
  k_cvt_wk<<<2304, 256, 0, stream>>>(qkv_w, Wk_h);
  k_qcls <<<3072, 256, 0, stream>>>(t_cls, qkv_w, qkv_b, q_cls);
  k_kgemm<<<dim3(37, 32), 256, 0, stream>>>(Th, Wk_h, qkv_b, K_rot);
  k_attn <<<dim3(NHd, Bn), 256, 0, stream>>>(q_cls, K_rot, x, t_cls, qkv_w, qkv_b, o_buf);
  k_proj <<<3072, 256, 0, stream>>>(o_buf, proj_w, proj_b, out);
}
